// TransformerDecoderBlock_58196806860819
// MI455X (gfx1250) — matrix operand order repaired and checked by reading, not compiled, not run on hardware
//
#include <hip/hip_runtime.h>
#include <hip/hip_bf16.h>

// ---------------------------------------------------------------------------
// CDNA5 (gfx1250, wave32) implementation of a linear-attention transformer
// decoder block. All GEMM-shaped math runs through v_wmma_f32_16x16x32_f16;
// f16 GEMM A-tiles are staged with GLOBAL_LOAD_ASYNC_TO_LDS_B128 (ASYNCcnt).
// ---------------------------------------------------------------------------

typedef __attribute__((ext_vector_type(16))) _Float16 v16h;
typedef __attribute__((ext_vector_type(8)))  float    v8f;
typedef __attribute__((ext_vector_type(4)))  unsigned int u32x4;

static constexpr int LQ   = 1024;   // decoder sequence length
static constexpr int SM   = 2048;   // memory sequence length
static constexpr int BB   = 16;     // batch
static constexpr int DD   = 512;    // model dim
static constexpr int HH   = 8;      // heads
static constexpr int DMLP = 2048;   // mlp hidden
static constexpr int MROW = LQ * BB;  // 16384 token rows
static constexpr int MMEM = SM * BB;  // 32768 memory rows

__device__ __forceinline__ v8f wmma_f16(v16h a, v16h b, v8f c) {
  // D = A(16x32 f16) * B(32x16 f16) + C(16x16 f32)
  return __builtin_amdgcn_wmma_f32_16x16x32_f16(
      /*neg_a=*/false, a, /*neg_b=*/false, b,
      /*c_mod=*/(short)0, c, /*reuse_a=*/false, /*reuse_b=*/false);
}

// Async DMA of 16 bytes global -> LDS (no VGPR data path, tracked by ASYNCcnt).
// LDS operand is the wave-relative LDS byte offset (= low 32 bits of the flat
// pointer for the LDS aperture, ISA 10.2).
__device__ __forceinline__ void async_ld_lds16(void* lds_dst, const void* gsrc) {
  unsigned loff = (unsigned)(uintptr_t)lds_dst;
  asm volatile("global_load_async_to_lds_b128 %0, %1, off"
               :: "v"(loff), "v"(gsrc)
               : "memory");
}
template <typename AT>
__device__ __forceinline__ void wait_stage() {}
template <>
__device__ __forceinline__ void wait_stage<_Float16>() {
  asm volatile("s_wait_asynccnt 0x0" ::: "memory");
}

// A fragment from a row-major (M x K) f16 LDS tile.
// ISA layout (16-bit A 16x32): lane r=lane&15 is row M=r; hi half-wave offsets
// K by 8; element e maps to K = hi*8 + e + (e>=8 ? 8 : 0).
// ld (in halves) must be a multiple of 8; k0 a multiple of 16.
__device__ __forceinline__ v16h ld_frag_a(const _Float16* m, int row0, int k0, int ld) {
  const int lane = threadIdx.x & 31;
  const int r = lane & 15, hi = lane >> 4;
  const _Float16* p = m + (row0 + r) * ld + k0 + (hi ? 8 : 0);
  v16h f;
  ((u32x4*)&f)[0] = *((const u32x4*)p);          // K = base + 0..7
  ((u32x4*)&f)[1] = *((const u32x4*)(p + 16));   // K = base + 16..23
  return f;
}

// B fragment (K x N) from an LDS tile stored N-major, i.e. m[n*ld + k].
// Layout: lane n=lane&15 is column; element e maps to K = k0 + hi*8 + e + (e>=8 ? 8 : 0).
__device__ __forceinline__ v16h ld_frag_bT(const _Float16* m, int k0, int col0, int ld) {
  const int lane = threadIdx.x & 31;
  const int n = lane & 15, hi = lane >> 4;
  const _Float16* p = m + (col0 + n) * ld + k0 + (hi ? 8 : 0);
  v16h f;
  ((u32x4*)&f)[0] = *((const u32x4*)p);          // K = base + 0..7
  ((u32x4*)&f)[1] = *((const u32x4*)(p + 16));   // K = base + 16..23
  return f;
}

// ---------------------------------------------------------------------------
// LayerNorm: one wave (32 lanes) per row of 512, writes f16 for WMMA consumers.
// ---------------------------------------------------------------------------
__global__ __launch_bounds__(256)
void ln_kernel(const float* __restrict__ x, const float* __restrict__ g,
               const float* __restrict__ b, _Float16* __restrict__ out, int M) {
  const int wave = threadIdx.x >> 5;
  const int lane = threadIdx.x & 31;
  const int row = blockIdx.x * 8 + wave;
  if (row >= M) return;
  const float* xr = x + (size_t)row * DD;
  float vals[16];
  float s = 0.f;
#pragma unroll
  for (int j = 0; j < 16; ++j) { vals[j] = xr[lane + 32 * j]; s += vals[j]; }
#pragma unroll
  for (int off = 16; off > 0; off >>= 1) s += __shfl_xor(s, off, 32);
  const float mean = s * (1.f / DD);
  float vs = 0.f;
#pragma unroll
  for (int j = 0; j < 16; ++j) { float d = vals[j] - mean; vs += d * d; }
#pragma unroll
  for (int off = 16; off > 0; off >>= 1) vs += __shfl_xor(vs, off, 32);
  const float inv = rsqrtf(vs * (1.f / DD) + 1e-5f);
  _Float16* orow = out + (size_t)row * DD;
#pragma unroll
  for (int j = 0; j < 16; ++j) {
    int c = lane + 32 * j;
    orow[c] = (_Float16)((vals[j] - mean) * inv * g[c] + b[c]);
  }
}

// rowscale[m] = mask[b * Lm + l] with m = l*B + b  (key-padding masks)
__global__ void rowscale_kernel(const float* __restrict__ mask,
                                float* __restrict__ rs, int Lm, int B) {
  int m = blockIdx.x * 256 + threadIdx.x;
  if (m >= Lm * B) return;
  int b = m % B, l = m / B;
  rs[m] = mask[(size_t)b * Lm + l];
}

// ---------------------------------------------------------------------------
// A-tile staging: f16 source goes through the async LDS DMA path; f32 source
// is converted to f16 on stage through VGPRs. `ap` points at the 16 elements
// this thread owns; `ls` is the matching LDS destination.
// ---------------------------------------------------------------------------
__device__ __forceinline__ void stage_a16(const _Float16* ap, _Float16* ls) {
  async_ld_lds16(ls, ap);            // halves 0..7
  async_ld_lds16(ls + 8, ap + 8);    // halves 8..15
}
__device__ __forceinline__ void stage_a16(const float* ap, _Float16* ls) {
#pragma unroll
  for (int e = 0; e < 16; ++e) ls[e] = (_Float16)ap[e];
}

// ---------------------------------------------------------------------------
// WMMA GEMM: O = epilogue(A[MxK] @ W[KxN] + bias), tile 128x128, 8 waves.
// A: f32 or f16 global (f16 path uses async global->LDS DMA).
// ACT: 0 = none, 1 = phi (elu+1), 2 = relu. Optional rowscale and residual.
// M,N,K all multiples of 128/128/32 here.
// ---------------------------------------------------------------------------
template <typename AT, typename OT, int ACT>
__global__ __launch_bounds__(256)
void gemm_kernel(const AT* __restrict__ A, const float* __restrict__ W,
                 const float* __restrict__ bias, const float* __restrict__ rowscale,
                 const float* __restrict__ residual, OT* __restrict__ O,
                 int M, int N, int K) {
  constexpr int BM = 128, BN = 128, BK = 32;
  constexpr int LDA = BK + 8;   // 40 halves = 80B rows (16B aligned)
  __shared__ __align__(16) _Float16 As[BM * LDA];   // row-major (m,k)
  __shared__ __align__(16) _Float16 BsT[BN * LDA];  // N-major  (n,k)

  const int m0 = blockIdx.y * BM, n0 = blockIdx.x * BN;
  const int tid = threadIdx.x;
  const int wave = tid >> 5, lane = tid & 31;
  const int wy = wave >> 1, wx = wave & 1;     // 4x2 wave grid
  const int wrow = wy * 32, wcol = wx * 64;

  v8f acc[2][4];
#pragma unroll
  for (int i = 0; i < 2; ++i)
#pragma unroll
    for (int j = 0; j < 4; ++j)
#pragma unroll
      for (int e = 0; e < 8; ++e) acc[i][j][e] = 0.f;

  const int ar = tid >> 1, ac = (tid & 1) * 16;          // A stage: 16 halves/thread
  const int kr = tid & 31, nb = (tid >> 5) * 16;         // W stage: 16 cols/thread

  for (int k0 = 0; k0 < K; k0 += BK) {
    // A tile: async DMA (f16) or convert-on-stage (f32), issued first so the
    // synchronous W staging below overlaps the in-flight copy.
    stage_a16(A + (size_t)(m0 + ar) * K + k0 + ac, &As[ar * LDA + ac]);
    const float* wp = W + (size_t)(k0 + kr) * N + n0 + nb;
#pragma unroll
    for (int e = 0; e < 16; ++e) BsT[(nb + e) * LDA + kr] = (_Float16)wp[e];
    if (k0 + BK < K)  // prefetch next W tile rows into GL2 (global_prefetch_b8)
      __builtin_prefetch(W + (size_t)(k0 + BK + kr) * N + n0 + nb, 0, 3);
    wait_stage<AT>();   // s_wait_asynccnt 0 on the f16 path
    __syncthreads();

    v16h af0 = ld_frag_a(As, wrow +  0, 0, LDA);
    v16h af1 = ld_frag_a(As, wrow + 16, 0, LDA);
#pragma unroll
    for (int j = 0; j < 4; ++j) {
      v16h bf = ld_frag_bT(BsT, 0, wcol + j * 16, LDA);
      acc[0][j] = wmma_f16(af0, bf, acc[0][j]);
      acc[1][j] = wmma_f16(af1, bf, acc[1][j]);
    }
    __syncthreads();
  }

  // Epilogue. C/D layout: lane n=lane&15 is column; vgpr r is row r + hi*8.
  const int nl = lane & 15, hi = lane >> 4;
#pragma unroll
  for (int i = 0; i < 2; ++i) {
#pragma unroll
    for (int j = 0; j < 4; ++j) {
      const int col = n0 + wcol + j * 16 + nl;
      const float bc = bias[col];
#pragma unroll
      for (int r = 0; r < 8; ++r) {
        const int row = m0 + wrow + i * 16 + r + hi * 8;
        float v = acc[i][j][r] + bc;
        if (ACT == 1) v = (v > 0.f) ? (v + 1.f) : __expf(v);   // elu(x)+1
        if (ACT == 2) v = fmaxf(v, 0.f);
        if (rowscale) v *= rowscale[row];
        if (residual) v += residual[(size_t)row * N + col];
        O[(size_t)row * N + col] = (OT)v;
      }
    }
  }
}

// ---------------------------------------------------------------------------
// Causal linear attention, chunkwise (C=64). One block (4 waves) per (b,h),
// sequential over 16 chunks. Per chunk, 4 WMMA GEMMs of 64x64x64:
//   A   = Q K^T (masked causal, rowsum -> denominator)
//   out = Q S_prev + mask(A) V
//   S  += K^T V ;  z += colsum(K)
// q/k/v/o are (M x 512) f16, column = h*64 + d, row m = l*B + b.
// ---------------------------------------------------------------------------
__global__ __launch_bounds__(128)
void self_attn_kernel(const _Float16* __restrict__ q, const _Float16* __restrict__ k,
                      const _Float16* __restrict__ v, _Float16* __restrict__ o,
                      int L, int B) {
  constexpr int LDA = 40, LDB = 72;
  __shared__ __align__(16) _Float16 Qs [64 * LDA];  // (c, dk) row-major
  __shared__ __align__(16) _Float16 Am [64 * LDA];  // masked QK^T (row, col)
  __shared__ __align__(16) _Float16 Ks [64 * LDB];  // (c, dk)  -> B^T source for QK^T
  __shared__ __align__(16) _Float16 KsT[64 * LDB];  // (dk, c)  -> A source for K^T V
  __shared__ __align__(16) _Float16 VsT[64 * LDB];  // (dv, c)  -> B^T source
  __shared__ __align__(16) _Float16 ShT[64 * LDB];  // (dv, kk) -> B^T of state
  __shared__ float Sf[64 * 65];                     // f32 state (kk, dv)
  __shared__ float zs[64];
  __shared__ float den[64];

  const int bh = blockIdx.x, b = bh / HH, h = bh % HH;
  const int tid = threadIdx.x, lane = tid & 31, wave = tid >> 5;
  const int nl = lane & 15, hi = lane >> 4;
  const int wrow = wave * 16;
  const size_t rowstride = (size_t)B * DD;
  const size_t base = (size_t)b * DD + h * 64;

  for (int i = tid; i < 64 * 65; i += 128) Sf[i] = 0.f;
  if (tid < 64) zs[tid] = 0.f;
  __syncthreads();

  const int nchunks = L / 64;
  for (int c = 0; c < nchunks; ++c) {
    // --- stage chunk: 2 threads per row, 32 cols each ---
    {
      const int r = tid >> 1, cb = (tid & 1) * 32;
      const size_t g = base + (size_t)(c * 64 + r) * rowstride + cb;
#pragma unroll 8
      for (int e = 0; e < 32; ++e) {
        _Float16 kv = k[g + e];
        Qs [r * LDA + cb + e] = q[g + e];
        Ks [r * LDB + cb + e] = kv;
        KsT[(cb + e) * LDB + r] = kv;
        VsT[(cb + e) * LDB + r] = v[g + e];
      }
    }
    if (tid < 64) den[tid] = 1e-6f;   // EPS
    __syncthreads();

    // f16 transposed snapshot of the state for the Q @ S_prev GEMM
    for (int i = tid; i < 4096; i += 128) {
      int kk = i >> 6, dv = i & 63;
      ShT[dv * LDB + kk] = (_Float16)Sf[kk * 65 + dv];
    }
    // den += q . z_prev
    {
      const int rr = tid >> 1, off = (tid & 1) * 32;
      float s = 0.f;
#pragma unroll 8
      for (int e = 0; e < 32; ++e) s += (float)Qs[rr * LDA + off + e] * zs[off + e];
      atomicAdd(&den[rr], s);
    }
    // A = Q K^T  (wave owns rows wrow..wrow+15, all 64 cols)
    v8f a4[4];
#pragma unroll
    for (int j = 0; j < 4; ++j) {
      v8f accq;
#pragma unroll
      for (int e = 0; e < 8; ++e) accq[e] = 0.f;
#pragma unroll
      for (int ks = 0; ks < 64; ks += 32)
        accq = wmma_f16(ld_frag_a(Qs, wrow, ks, LDA), ld_frag_bT(Ks, ks, j * 16, LDB), accq);
      a4[j] = accq;
    }
    // causal mask + denominator rowsums + store Am (f16) for second GEMM
#pragma unroll
    for (int j = 0; j < 4; ++j) {
      const int colc = j * 16 + nl;
#pragma unroll
      for (int r = 0; r < 8; ++r) {
        const int rowc = wrow + r + hi * 8;
        const float val = (colc <= rowc) ? a4[j][r] : 0.f;
        Am[rowc * LDA + colc] = (_Float16)val;
        atomicAdd(&den[rowc], val);
      }
    }
    __syncthreads();   // Am, ShT, den complete

    // out = Q @ S_prev + mask(A) @ V
    v8f oacc[4];
#pragma unroll
    for (int j = 0; j < 4; ++j) {
      v8f accq;
#pragma unroll
      for (int e = 0; e < 8; ++e) accq[e] = 0.f;
#pragma unroll
      for (int ks = 0; ks < 64; ks += 32)
        accq = wmma_f16(ld_frag_a(Qs, wrow, ks, LDA), ld_frag_bT(ShT, ks, j * 16, LDB), accq);
#pragma unroll
      for (int ks = 0; ks < 64; ks += 32)
        accq = wmma_f16(ld_frag_a(Am, wrow, ks, LDA), ld_frag_bT(VsT, ks, j * 16, LDB), accq);
      oacc[j] = accq;
    }
    // S += K^T V  (wave owns S rows (dk) wrow..wrow+15)
#pragma unroll
    for (int j = 0; j < 4; ++j) {
      v8f accq;
#pragma unroll
      for (int e = 0; e < 8; ++e) accq[e] = 0.f;
#pragma unroll
      for (int ks = 0; ks < 64; ks += 32)
        accq = wmma_f16(ld_frag_a(KsT, wrow, ks, LDB), ld_frag_bT(VsT, ks, j * 16, LDB), accq);
#pragma unroll
      for (int r = 0; r < 8; ++r) {
        const int dk = wrow + r + hi * 8, dv = j * 16 + nl;
        Sf[dk * 65 + dv] += accq[r];
      }
    }
    // z += colsum(K)
    {
      const int dk = tid >> 1, off = (tid & 1) * 32;
      float s = 0.f;
#pragma unroll 8
      for (int e = 0; e < 32; ++e) s += (float)KsT[dk * LDB + off + e];
      atomicAdd(&zs[dk], s);
    }
    // store out / den
#pragma unroll
    for (int j = 0; j < 4; ++j) {
#pragma unroll
      for (int r = 0; r < 8; ++r) {
        const int rowc = wrow + r + hi * 8;
        const int colc = j * 16 + nl;
        const size_t go = base + (size_t)(c * 64 + rowc) * rowstride + colc;
        o[go] = (_Float16)(oacc[j][r] / den[rowc]);
      }
    }
    __syncthreads();
  }
}

// ---------------------------------------------------------------------------
// Cross-attention global reduction: KV = K^T V (64x64), z = colsum(K),
// per (b,h). Accumulators live in registers across all 32 memory chunks.
// ---------------------------------------------------------------------------
__global__ __launch_bounds__(128)
void cross_kv_kernel(const _Float16* __restrict__ k, const _Float16* __restrict__ v,
                     _Float16* __restrict__ KV, float* __restrict__ z, int S, int B) {
  constexpr int LDB = 72;
  __shared__ __align__(16) _Float16 KsT[64 * LDB];  // (dk, c)
  __shared__ __align__(16) _Float16 VsT[64 * LDB];  // (dv, c)
  __shared__ float zs[64];
  const int bh = blockIdx.x, b = bh / HH, h = bh % HH;
  const int tid = threadIdx.x, lane = tid & 31, wave = tid >> 5;
  const int wrow = wave * 16;
  const size_t rowstride = (size_t)B * DD;
  const size_t base = (size_t)b * DD + h * 64;
  if (tid < 64) zs[tid] = 0.f;

  v8f acc[4];
#pragma unroll
  for (int j = 0; j < 4; ++j)
#pragma unroll
    for (int e = 0; e < 8; ++e) acc[j][e] = 0.f;

  for (int c = 0; c < S / 64; ++c) {
    __syncthreads();
    const int r = tid >> 1, cb = (tid & 1) * 32;
    const size_t g = base + (size_t)(c * 64 + r) * rowstride + cb;
#pragma unroll 8
    for (int e = 0; e < 32; ++e) {
      KsT[(cb + e) * LDB + r] = k[g + e];
      VsT[(cb + e) * LDB + r] = v[g + e];
    }
    __syncthreads();
#pragma unroll
    for (int j = 0; j < 4; ++j)
#pragma unroll
      for (int ks = 0; ks < 64; ks += 32)
        acc[j] = wmma_f16(ld_frag_a(KsT, wrow, ks, LDB), ld_frag_bT(VsT, ks, j * 16, LDB), acc[j]);
    const int dk = tid >> 1, off = (tid & 1) * 32;
    float s = 0.f;
#pragma unroll 8
    for (int e = 0; e < 32; ++e) s += (float)KsT[dk * LDB + off + e];
    atomicAdd(&zs[dk], s);
  }
  __syncthreads();
  const int nl = lane & 15, hi = lane >> 4;
  const size_t ob = (size_t)bh * 64 * 64;
#pragma unroll
  for (int j = 0; j < 4; ++j)
#pragma unroll
    for (int r = 0; r < 8; ++r) {
      const int dk = wrow + r + hi * 8, dv = j * 16 + nl;
      KV[ob + dk * 64 + dv] = (_Float16)acc[j][r];
    }
  if (tid < 64) z[(size_t)bh * 64 + tid] = zs[tid];
}

// num = Q @ KV ; den = q.z + EPS ; out = num/den. Block = (bh, l-chunk of 64).
__global__ __launch_bounds__(128)
void cross_apply_kernel(const _Float16* __restrict__ q, const _Float16* __restrict__ KV,
                        const float* __restrict__ z, _Float16* __restrict__ o,
                        int L, int B) {
  constexpr int LDA = 40, LDB = 72;
  __shared__ __align__(16) _Float16 Qs [64 * LDA];  // (l, dk)
  __shared__ __align__(16) _Float16 KVT[64 * LDB];  // (dv, dk)
  __shared__ float zs[64];
  __shared__ float den[64];
  const int bh = blockIdx.x, b = bh / HH, h = bh % HH;
  const int lc = blockIdx.y;
  const int tid = threadIdx.x, lane = tid & 31, wave = tid >> 5;
  const int wrow = wave * 16;
  const size_t rowstride = (size_t)B * DD;
  const size_t base = (size_t)b * DD + h * 64;

  for (int i = tid; i < 4096; i += 128) {
    int dk = i >> 6, dv = i & 63;
    KVT[dv * LDB + dk] = KV[(size_t)bh * 4096 + i];
  }
  if (tid < 64) { zs[tid] = z[(size_t)bh * 64 + tid]; den[tid] = 1e-6f; }
  {
    const int r = tid >> 1, cb = (tid & 1) * 32;
    const size_t g = base + (size_t)(lc * 64 + r) * rowstride + cb;
#pragma unroll 8
    for (int e = 0; e < 32; ++e) Qs[r * LDA + cb + e] = q[g + e];
  }
  __syncthreads();
  {
    const int rr = tid >> 1, off = (tid & 1) * 32;
    float s = 0.f;
#pragma unroll 8
    for (int e = 0; e < 32; ++e) s += (float)Qs[rr * LDA + off + e] * zs[off + e];
    atomicAdd(&den[rr], s);
  }
  v8f acc[4];
#pragma unroll
  for (int j = 0; j < 4; ++j) {
#pragma unroll
    for (int e = 0; e < 8; ++e) acc[j][e] = 0.f;
#pragma unroll
    for (int ks = 0; ks < 64; ks += 32)
      acc[j] = wmma_f16(ld_frag_a(Qs, wrow, ks, LDA), ld_frag_bT(KVT, ks, j * 16, LDB), acc[j]);
  }
  __syncthreads();
  const int nl = lane & 15, hi = lane >> 4;
#pragma unroll
  for (int j = 0; j < 4; ++j)
#pragma unroll
    for (int r = 0; r < 8; ++r) {
      const int rowc = wrow + r + hi * 8;
      const size_t go = base + (size_t)(lc * 64 + rowc) * rowstride + j * 16 + nl;
      o[go] = (_Float16)(acc[j][r] / den[rowc]);
    }
}

// ---------------------------------------------------------------------------
// Host-side launch sequence (graph-capture safe: only kernel launches).
// ---------------------------------------------------------------------------
extern "C" void kernel_launch(void* const* d_in, const int* in_sizes, int n_in,
                              void* d_out, int out_size, void* d_ws, size_t ws_size,
                              hipStream_t stream) {
  (void)in_sizes; (void)n_in; (void)out_size; (void)ws_size;
  const float* x      = (const float*)d_in[0];
  const float* memory = (const float*)d_in[1];
  const float* imask  = (const float*)d_in[2];
  const float* mmask  = (const float*)d_in[3];
  const float* ln1g = (const float*)d_in[4],  *ln1b = (const float*)d_in[5];
  const float* ln2g = (const float*)d_in[6],  *ln2b = (const float*)d_in[7];
  const float* ln3g = (const float*)d_in[8],  *ln3b = (const float*)d_in[9];
  const float* sa_wq = (const float*)d_in[10], *sa_bq = (const float*)d_in[11];
  const float* sa_wk = (const float*)d_in[12], *sa_bk = (const float*)d_in[13];
  const float* sa_wv = (const float*)d_in[14], *sa_bv = (const float*)d_in[15];
  const float* sa_wo = (const float*)d_in[16], *sa_bo = (const float*)d_in[17];
  const float* ca_wq = (const float*)d_in[18], *ca_bq = (const float*)d_in[19];
  const float* ca_wk = (const float*)d_in[20], *ca_bk = (const float*)d_in[21];
  const float* ca_wv = (const float*)d_in[22], *ca_bv = (const float*)d_in[23];
  const float* ca_wo = (const float*)d_in[24], *ca_bo = (const float*)d_in[25];
  const float* mw1 = (const float*)d_in[26], *mb1 = (const float*)d_in[27];
  const float* mw2 = (const float*)d_in[28], *mb2 = (const float*)d_in[29];
  float* out = (float*)d_out;

  // workspace carve (256B aligned)
  char* w = (char*)d_ws;
  auto carve = [&](size_t bytes) -> void* {
    void* p = (void*)w; w += (bytes + 255) & ~(size_t)255; return p;
  };
  _Float16* h16  = (_Float16*)carve((size_t)MROW * DD * 2);    // LN output
  _Float16* q16  = (_Float16*)carve((size_t)MROW * DD * 2);    // q / cross-q
  _Float16* k16  = (_Float16*)carve((size_t)MMEM * DD * 2);    // k / cross-k
  _Float16* v16  = (_Float16*)carve((size_t)MMEM * DD * 2);    // v / cross-v
  _Float16* ao16 = (_Float16*)carve((size_t)MROW * DD * 2);    // attn output
  float*    x1   = (float*)carve((size_t)MROW * DD * 4);
  float*    x2   = (float*)carve((size_t)MROW * DD * 4);
  _Float16* mh16 = (_Float16*)carve((size_t)MROW * DMLP * 2);  // mlp hidden
  _Float16* kv16 = (_Float16*)carve((size_t)BB * HH * 64 * 64 * 2);
  float*    zbuf = (float*)carve((size_t)BB * HH * 64 * 4);
  float*    rs_q = (float*)carve((size_t)MROW * 4);
  float*    rs_m = (float*)carve((size_t)MMEM * 4);

  const dim3 blk256(256), blk128(128);
  const dim3 gD (DD   / 128, MROW / 128);   // (4, 128)
  const dim3 gDs(DD   / 128, MMEM / 128);   // (4, 256)
  const dim3 gM1(DMLP / 128, MROW / 128);   // (16, 128)

  rowscale_kernel<<<(MROW + 255) / 256, blk256, 0, stream>>>(imask, rs_q, LQ, BB);
  rowscale_kernel<<<(MMEM + 255) / 256, blk256, 0, stream>>>(mmask, rs_m, SM, BB);

  // ---- self attention ----
  ln_kernel<<<MROW / 8, blk256, 0, stream>>>(x, ln1g, ln1b, h16, MROW);
  gemm_kernel<_Float16, _Float16, 1><<<gD, blk256, 0, stream>>>(h16, sa_wq, sa_bq, nullptr, nullptr, q16, MROW, DD, DD);
  gemm_kernel<_Float16, _Float16, 1><<<gD, blk256, 0, stream>>>(h16, sa_wk, sa_bk, rs_q,    nullptr, k16, MROW, DD, DD);
  gemm_kernel<_Float16, _Float16, 0><<<gD, blk256, 0, stream>>>(h16, sa_wv, sa_bv, nullptr, nullptr, v16, MROW, DD, DD);
  self_attn_kernel<<<BB * HH, blk128, 0, stream>>>(q16, k16, v16, ao16, LQ, BB);
  gemm_kernel<_Float16, float, 0><<<gD, blk256, 0, stream>>>(ao16, sa_wo, sa_bo, nullptr, x, x1, MROW, DD, DD);

  // ---- cross attention ----
  ln_kernel<<<MROW / 8, blk256, 0, stream>>>(x1, ln2g, ln2b, h16, MROW);
  gemm_kernel<_Float16, _Float16, 1><<<gD,  blk256, 0, stream>>>(h16,    ca_wq, ca_bq, nullptr, nullptr, q16, MROW, DD, DD);
  gemm_kernel<float,    _Float16, 1><<<gDs, blk256, 0, stream>>>(memory, ca_wk, ca_bk, rs_m,    nullptr, k16, MMEM, DD, DD);
  gemm_kernel<float,    _Float16, 0><<<gDs, blk256, 0, stream>>>(memory, ca_wv, ca_bv, nullptr, nullptr, v16, MMEM, DD, DD);
  cross_kv_kernel<<<BB * HH, blk128, 0, stream>>>(k16, v16, kv16, zbuf, SM, BB);
  cross_apply_kernel<<<dim3(BB * HH, LQ / 64), blk128, 0, stream>>>(q16, kv16, zbuf, ao16, LQ, BB);
  gemm_kernel<_Float16, float, 0><<<gD, blk256, 0, stream>>>(ao16, ca_wo, ca_bo, nullptr, x1, x2, MROW, DD, DD);

  // ---- MLP ----
  ln_kernel<<<MROW / 8, blk256, 0, stream>>>(x2, ln3g, ln3b, h16, MROW);
  gemm_kernel<_Float16, _Float16, 2><<<gM1, blk256, 0, stream>>>(h16,  mw1, mb1, nullptr, nullptr, mh16, MROW, DMLP, DD);
  gemm_kernel<_Float16, float,    0><<<gD,  blk256, 0, stream>>>(mh16, mw2, mb2, nullptr, x2, out, MROW, DD, DMLP);
}
